// MPCrystalGraphConvNet_66881230733570
// MI455X (gfx1250) — matrix operand order repaired and checked by reading, not compiled, not run on hardware
//
#include <hip/hip_runtime.h>
#include <cstdint>

// ---------------- problem constants ----------------
#define NATOMS 32768
#define MNBR   12
#define NMROWS (NATOMS * MNBR)     // 393216
#define ORIGF  92
#define NBRF   41
#define AF     64
#define HF     128
#define NCONV  3
#define APERC  32
#define NCRYS  (NATOMS / APERC)    // 1024
#define EPSBN  1e-5f

typedef __attribute__((ext_vector_type(16))) _Float16 v16h;
typedef __attribute__((ext_vector_type(8)))  float    v8f;

// ---------------- device helpers ----------------
__device__ __forceinline__ float softplusf(float x) {
    return fmaxf(x, 0.f) + log1pf(expf(-fabsf(x)));
}
__device__ __forceinline__ float sigmoidf(float x) {
    return 1.f / (1.f + expf(-x));
}

// Load a 16x32 f16 A-fragment from row-major memory.
// p points at (tile_row0, k_chunk*32); ld is the row stride in halves.
// Lane L: row = L&15; VGPR0-3 hold K = 8*(L>>4)+0..7, VGPR4-7 hold K = 16+8*(L>>4)+0..7.
__device__ __forceinline__ v16h load_a_frag(const _Float16* __restrict__ p, int ld) {
    int lane = threadIdx.x & 31;
    const _Float16* q = p + (size_t)(lane & 15) * ld + ((lane >> 4) << 3);
    union { v16h h; uint4 u[2]; } t;
    t.u[0] = *reinterpret_cast<const uint4*>(q);
    t.u[1] = *reinterpret_cast<const uint4*>(q + 16);
    return t.h;
}

// Load a 32x16 B-fragment from LDS (fragment-major layout staged by the block).
__device__ __forceinline__ v16h load_b_frag_lds(const uint4* sh, int frag) {
    int lane = threadIdx.x & 31;
    const uint4* q = sh + ((size_t)frag * 32 + lane) * 2;
    union { v16h h; uint4 u[2]; } t;
    t.u[0] = q[0];
    t.u[1] = q[1];
    return t.h;
}

__device__ __forceinline__ v8f wmma_f16(v16h a, v16h b, v8f c) {
    return __builtin_amdgcn_wmma_f32_16x16x32_f16(false, a, false, b, (short)0, c, false, false);
}

// ---------------- weight packing ----------------
// Pack row-major f32 W (ld columns) into per-fragment f16 layout:
// dst[((kt*Ntiles+nt)*32 + lane)*8 + d] holds K = kt*32 + (lane>>4)*8 + 2d + (d>=4?8:0), +1
// at column n = nt*16 + (lane&15). K >= Kvalid rows are zero.
__global__ void k_pack(const float* __restrict__ W, int ldw, int Kvalid,
                       int Ktiles, int Ntiles, unsigned int* __restrict__ dst) {
    int t = blockIdx.x * blockDim.x + threadIdx.x;
    int total = Ktiles * Ntiles * 32;
    if (t >= total) return;
    int lane = t & 31;
    int nt   = (t >> 5) % Ntiles;
    int kt   = t / (32 * Ntiles);
    int n     = nt * 16 + (lane & 15);
    int kbase = kt * 32 + ((lane >> 4) << 3);
    unsigned int* o = dst + ((size_t)(kt * Ntiles + nt) * 32 + lane) * 8;
    #pragma unroll
    for (int d = 0; d < 8; ++d) {
        int k0 = kbase + 2 * d + ((d >> 2) << 3), k1 = k0 + 1;
        union { _Float16 h; unsigned short s; } a, b;
        a.h = (k0 < Kvalid) ? (_Float16)W[(size_t)k0 * ldw + n] : (_Float16)0.f;
        b.h = (k1 < Kvalid) ? (_Float16)W[(size_t)k1 * ldw + n] : (_Float16)0.f;
        o[d] = (unsigned int)a.s | ((unsigned int)b.s << 16);
    }
}

// ---------------- dtype conversion ----------------
__global__ void k_cvt_pad(const float* __restrict__ src, _Float16* __restrict__ dst,
                          int cols_src, int cols_dst, long long total) {
    long long t = (long long)blockIdx.x * blockDim.x + threadIdx.x;
    if (t >= total) return;
    long long r = t / cols_dst;
    int c = (int)(t - r * cols_dst);
    float v = (c < cols_src) ? src[r * cols_src + c] : 0.f;
    dst[t] = (_Float16)v;
}

// ---------------- generic WMMA GEMM ----------------
// One wave computes a 16 x (NT*16) output tile; 8 waves per block.
// B fragments are staged once per block into LDS (shared by all 8 waves).
// ACT: 0 = none, 1 = softplus.
template<int KT, int NT, int ACT, bool OF32, bool OF16>
__global__ __launch_bounds__(256)
void k_gemm(const _Float16* __restrict__ A, int lda,
            const unsigned int* __restrict__ Bpk,
            const float* __restrict__ bias,
            float* __restrict__ Of, _Float16* __restrict__ Oh, int ldo) {
    __shared__ uint4 Bsh[KT * NT * 64];
    for (int i = threadIdx.x; i < KT * NT * 64; i += 256)
        Bsh[i] = reinterpret_cast<const uint4*>(Bpk)[i];
    __syncthreads();

    int wave = threadIdx.x >> 5;
    int r0 = (blockIdx.x * 8 + wave) * 16;
    v8f c[NT] = {};
    #pragma unroll
    for (int kt = 0; kt < KT; ++kt) {
        v16h a = load_a_frag(A + (size_t)r0 * lda + kt * 32, lda);
        #pragma unroll
        for (int nt = 0; nt < NT; ++nt) {
            v16h b = load_b_frag_lds(Bsh, kt * NT + nt);
            c[nt] = wmma_f16(a, b, c[nt]);
        }
    }
    int lane = threadIdx.x & 31;
    int col0 = lane & 15;
    int radd = (lane >> 4) * 8;
    #pragma unroll
    for (int nt = 0; nt < NT; ++nt) {
        int n = nt * 16 + col0;
        float bb = bias[n];
        #pragma unroll
        for (int v = 0; v < 8; ++v) {
            int r = r0 + radd + v;
            float val = c[nt][v] + bb;
            if (ACT == 1) val = softplusf(val);
            if (OF32) Of[(size_t)r * ldo + n] = val;
            if (OF16) Oh[(size_t)r * ldo + n] = (_Float16)val;
        }
    }
}

// ---------------- conv message GEMM (the big one) ----------------
// Rows j in [0, N*M): A row = [ x_f16[nbr_idx[j]] (64) | nbr_pad_f16[j] (64, zero-padded from 41) ]
// B = packed conv_W rows 64..168 (K padded to 128), staged in LDS per block.
// Epilogue adds S[atom] (self-term + bias), stores g in f16, and accumulates
// per-channel sum/sumsq for BN1 via f32 atomics.
__global__ __launch_bounds__(256)
void k_conv(const _Float16* __restrict__ x16,
            const _Float16* __restrict__ nbr16,
            const int* __restrict__ nbr_idx,
            const unsigned int* __restrict__ Bpk,
            const float* __restrict__ S,
            _Float16* __restrict__ g16,
            float* __restrict__ bn1_sum, float* __restrict__ bn1_sq) {
    __shared__ uint4 Bsh[4 * 8 * 64];          // 32 KB: all 32 B-fragments
    for (int i = threadIdx.x; i < 4 * 8 * 64; i += 256)
        Bsh[i] = reinterpret_cast<const uint4*>(Bpk)[i];

    int wave = threadIdx.x >> 5;
    int j0 = (blockIdx.x * 8 + wave) * 16;
    int lane = threadIdx.x & 31;
    int row = lane & 15;

    // Prefetch epilogue self-term rows (consumed after the wmma chain drains).
    __builtin_prefetch(&S[(size_t)((j0 + row) / MNBR) * (2 * AF) + ((lane >> 4) << 6)], 0, 3);

    // Gathered-atom A fragments: issue the index + row loads before the barrier.
    int idx = nbr_idx[j0 + row];
    const _Float16* arow = x16 + (size_t)idx * AF + ((lane >> 4) << 3);
    union { v16h h; uint4 u[2]; } a01[2];
    #pragma unroll
    for (int kt = 0; kt < 2; ++kt) {
        const _Float16* q = arow + kt * 32;
        a01[kt].u[0] = *reinterpret_cast<const uint4*>(q);
        a01[kt].u[1] = *reinterpret_cast<const uint4*>(q + 16);
    }
    __syncthreads();

    v8f c[8] = {};
    // K chunks 0..1: gathered atom features
    #pragma unroll
    for (int kt = 0; kt < 2; ++kt) {
        #pragma unroll
        for (int nt = 0; nt < 8; ++nt) {
            v16h b = load_b_frag_lds(Bsh, kt * 8 + nt);
            c[nt] = wmma_f16(a01[kt].h, b, c[nt]);
        }
    }
    // K chunks 2..3: padded neighbor edge features
    #pragma unroll
    for (int kt = 2; kt < 4; ++kt) {
        v16h a = load_a_frag(nbr16 + (size_t)j0 * AF + (kt - 2) * 32, AF);
        #pragma unroll
        for (int nt = 0; nt < 8; ++nt) {
            v16h b = load_b_frag_lds(Bsh, kt * 8 + nt);
            c[nt] = wmma_f16(a, b, c[nt]);
        }
    }

    int col0 = lane & 15;
    int radd = (lane >> 4) * 8;
    #pragma unroll
    for (int nt = 0; nt < 8; ++nt) {
        int n = nt * 16 + col0;
        float s = 0.f, q = 0.f;
        #pragma unroll
        for (int v = 0; v < 8; ++v) {
            int j = j0 + radd + v;
            int atom = j / MNBR;
            float val = c[nt][v] + S[(size_t)atom * (2 * AF) + n];
            g16[(size_t)j * (2 * AF) + n] = (_Float16)val;
            s += val;
            q += val * val;
        }
        atomicAdd(&bn1_sum[n], s);
        atomicAdd(&bn1_sq[n], q);
    }
}

// ---------------- BN finalize (scale/shift from sum/sumsq) ----------------
__global__ void k_bn_final(const float* __restrict__ sum, const float* __restrict__ sq,
                           const float* __restrict__ g, const float* __restrict__ b,
                           float cnt, float* __restrict__ scale, float* __restrict__ shift) {
    int c = threadIdx.x;
    float mean = sum[c] / cnt;
    float var  = sq[c] / cnt - mean * mean;
    float sc = g[c] * rsqrtf(var + EPSBN);
    scale[c] = sc;
    shift[c] = b[c] - mean * sc;
}

// ---------------- gate + neighbor sum + BN2 stats ----------------
// 256 threads = 4 atoms x 64 channels. summed[i,c] = sum_m sigmoid(gn[..c])*softplus(gn[..c+64])
__global__ __launch_bounds__(256)
void k_gate(const _Float16* __restrict__ g16,
            const float* __restrict__ sc, const float* __restrict__ sh,
            float* __restrict__ summed,
            float* __restrict__ bn2_sum, float* __restrict__ bn2_sq) {
    __shared__ float ls[AF], lq[AF];
    int t = threadIdx.x;
    if (t < AF) { ls[t] = 0.f; lq[t] = 0.f; }
    __syncthreads();
    int ci = t & 63;
    int i  = blockIdx.x * 4 + (t >> 6);
    float scF = sc[ci], shF = sh[ci], scC = sc[AF + ci], shC = sh[AF + ci];
    float acc = 0.f;
    #pragma unroll
    for (int m = 0; m < MNBR; ++m) {
        size_t base = ((size_t)i * MNBR + m) * (2 * AF);
        float gf = (float)g16[base + ci]      * scF + shF;
        float gc = (float)g16[base + AF + ci] * scC + shC;
        acc += sigmoidf(gf) * softplusf(gc);
    }
    summed[(size_t)i * AF + ci] = acc;
    atomicAdd(&ls[ci], acc);
    atomicAdd(&lq[ci], acc * acc);
    __syncthreads();
    if (t < AF) { atomicAdd(&bn2_sum[t], ls[t]); atomicAdd(&bn2_sq[t], lq[t]); }
}

// ---------------- residual update: x = softplus(x + BN2(summed)) ----------------
__global__ void k_update(const float* __restrict__ summed,
                         const float* __restrict__ sc2, const float* __restrict__ sh2,
                         float* __restrict__ x, _Float16* __restrict__ x16) {
    int t = blockIdx.x * blockDim.x + threadIdx.x;
    int c = t & 63;
    float v = summed[t] * sc2[c] + sh2[c];
    float nv = softplusf(x[t] + v);
    x[t] = nv;
    x16[t] = (_Float16)nv;
}

// ---------------- crystal mean pool + softplus ----------------
__global__ void k_pool(const float* __restrict__ x, _Float16* __restrict__ h16) {
    int t = blockIdx.x * blockDim.x + threadIdx.x;   // over NCRYS*AF
    int b = t >> 6, c = t & 63;
    float s = 0.f;
    #pragma unroll 4
    for (int a = 0; a < APERC; ++a)
        s += x[(size_t)(b * APERC + a) * AF + c];
    h16[t] = (_Float16)softplusf(s * (1.f / APERC));
}

// ---------------- output head: dot(h2, out_W) + out_b ----------------
__global__ void k_out(const float* __restrict__ h2, const float* __restrict__ ow,
                      const float* __restrict__ ob, float* __restrict__ out) {
    int b = blockIdx.x * blockDim.x + threadIdx.x;
    float acc = 0.f;
    #pragma unroll 8
    for (int c = 0; c < HF; ++c) acc += h2[(size_t)b * HF + c] * ow[c];
    out[b] = acc + ob[0];
}

// =============================================================================
extern "C" void kernel_launch(void* const* d_in, const int* in_sizes, int n_in,
                              void* d_out, int out_size, void* d_ws, size_t ws_size,
                              hipStream_t stream) {
    const float* atom_fea = (const float*)d_in[0];
    const float* nbr_fea  = (const float*)d_in[1];
    const int*   nbr_idx  = (const int*)d_in[2];
    // d_in[3] crystal_id: equal 32-atom blocks, structure used directly
    const float* emb_W = (const float*)d_in[4];
    const float* emb_b = (const float*)d_in[5];
    const float* conv_W = (const float*)d_in[6];
    const float* conv_b = (const float*)d_in[7];
    const float* bn1_g = (const float*)d_in[8];
    const float* bn1_b = (const float*)d_in[9];
    const float* bn2_g = (const float*)d_in[10];
    const float* bn2_b = (const float*)d_in[11];
    const float* fc1_W = (const float*)d_in[12];
    const float* fc1_b = (const float*)d_in[13];
    const float* fc2_W = (const float*)d_in[14];
    const float* fc2_b = (const float*)d_in[15];
    const float* out_W = (const float*)d_in[16];
    const float* out_b = (const float*)d_in[17];
    float* out = (float*)d_out;

    // ---- workspace carve-up (256B aligned) ----
    char* base = (char*)d_ws;
    size_t off = 0;
    auto carve = [&](size_t bytes) -> char* {
        char* p = base + off;
        off = (off + bytes + 255) & ~(size_t)255;
        return p;
    };
    _Float16* atom16 = (_Float16*)carve((size_t)NATOMS * 96 * 2);       // padded 92->96
    _Float16* nbr16  = (_Float16*)carve((size_t)NMROWS * AF * 2);       // padded 41->64
    float*    x      = (float*)   carve((size_t)NATOMS * AF * 4);
    _Float16* x16    = (_Float16*)carve((size_t)NATOMS * AF * 2);
    float*    S      = (float*)   carve((size_t)NATOMS * 2 * AF * 4);   // self-term + bias
    _Float16* g16    = (_Float16*)carve((size_t)NMROWS * 2 * AF * 2);
    float*    summed = (float*)   carve((size_t)NATOMS * AF * 4);
    float*    stats  = (float*)   carve(768 * 4);
    float *bn1_sum = stats,        *bn1_sq = stats + 128;
    float *bn2_sum = stats + 256,  *bn2_sq = stats + 320;
    float *bn1_sc  = stats + 384,  *bn1_sh = stats + 512;
    float *bn2_sc  = stats + 640,  *bn2_sh = stats + 704;
    _Float16* h16  = (_Float16*)carve((size_t)NCRYS * AF * 2);
    _Float16* h116 = (_Float16*)carve((size_t)NCRYS * HF * 2);
    float*    h2   = (float*)   carve((size_t)NCRYS * HF * 4);
    unsigned int* wEmb  = (unsigned int*)carve((size_t)3 * 4 * 32 * 8 * 4);
    unsigned int* wSelf[NCONV];
    unsigned int* wMsg[NCONV];
    for (int l = 0; l < NCONV; ++l) {
        wSelf[l] = (unsigned int*)carve((size_t)2 * 8 * 32 * 8 * 4);
        wMsg[l]  = (unsigned int*)carve((size_t)4 * 8 * 32 * 8 * 4);
    }
    unsigned int* wFc1 = (unsigned int*)carve((size_t)2 * 8 * 32 * 8 * 4);
    unsigned int* wFc2 = (unsigned int*)carve((size_t)4 * 8 * 32 * 8 * 4);

    // ---- pack all weights into B-fragment layout ----
    k_pack<<<2, 256, 0, stream>>>(emb_W, AF, ORIGF, 3, 4, wEmb);
    for (int l = 0; l < NCONV; ++l) {
        const float* Wl = conv_W + (size_t)l * (2 * AF + NBRF) * (2 * AF);
        // self part: rows 0..63
        k_pack<<<2, 256, 0, stream>>>(Wl, 2 * AF, AF, 2, 8, wSelf[l]);
        // message part kt 0..1: rows 64..127 (gathered-atom K slice)
        k_pack<<<2, 256, 0, stream>>>(Wl + (size_t)AF * 2 * AF, 2 * AF, AF, 2, 8, wMsg[l]);
        // message part kt 2..3: rows 128..168 (edge-feature K slice, zero-padded)
        k_pack<<<2, 256, 0, stream>>>(Wl + (size_t)2 * AF * 2 * AF, 2 * AF, NBRF, 2, 8,
                                      wMsg[l] + (size_t)2 * 8 * 32 * 8);
    }
    k_pack<<<2, 256, 0, stream>>>(fc1_W, HF, AF, 2, 8, wFc1);
    k_pack<<<4, 256, 0, stream>>>(fc2_W, HF, HF, 4, 8, wFc2);

    // ---- f16 conversions with K padding ----
    {
        long long tot = (long long)NATOMS * 96;
        k_cvt_pad<<<(unsigned)((tot + 255) / 256), 256, 0, stream>>>(atom_fea, atom16, ORIGF, 96, tot);
        tot = (long long)NMROWS * AF;
        k_cvt_pad<<<(unsigned)((tot + 255) / 256), 256, 0, stream>>>(nbr_fea, nbr16, NBRF, AF, tot);
    }

    // ---- embedding: x = atom_fea @ emb_W + emb_b ----
    k_gemm<3, 4, 0, true, true><<<NATOMS / 128, 256, 0, stream>>>(
        atom16, 96, wEmb, emb_b, x, x16, AF);

    // ---- conv layers ----
    for (int l = 0; l < NCONV; ++l) {
        hipMemsetAsync(stats, 0, 384 * sizeof(float), stream);
        // self-term + bias: S = x @ W[0:64] + conv_b
        k_gemm<2, 8, 0, true, false><<<NATOMS / 128, 256, 0, stream>>>(
            x16, AF, wSelf[l], conv_b + (size_t)l * 2 * AF, S, nullptr, 2 * AF);
        // message GEMM with fused gather + BN1 stats
        k_conv<<<NMROWS / 128, 256, 0, stream>>>(
            x16, nbr16, nbr_idx, wMsg[l], S, g16, bn1_sum, bn1_sq);
        k_bn_final<<<1, 128, 0, stream>>>(bn1_sum, bn1_sq,
            bn1_g + (size_t)l * 2 * AF, bn1_b + (size_t)l * 2 * AF,
            (float)NMROWS, bn1_sc, bn1_sh);
        k_gate<<<NATOMS / 4, 256, 0, stream>>>(g16, bn1_sc, bn1_sh, summed, bn2_sum, bn2_sq);
        k_bn_final<<<1, 64, 0, stream>>>(bn2_sum, bn2_sq,
            bn2_g + (size_t)l * AF, bn2_b + (size_t)l * AF,
            (float)NATOMS, bn2_sc, bn2_sh);
        k_update<<<(NATOMS * AF) / 256, 256, 0, stream>>>(summed, bn2_sc, bn2_sh, x, x16);
    }

    // ---- head ----
    k_pool<<<(NCRYS * AF) / 256, 256, 0, stream>>>(x, h16);
    k_gemm<2, 8, 1, false, true><<<NCRYS / 128, 256, 0, stream>>>(
        h16, AF, wFc1, fc1_b, nullptr, h116, HF);
    k_gemm<4, 8, 1, true, false><<<NCRYS / 128, 256, 0, stream>>>(
        h116, HF, wFc2, fc2_b, h2, nullptr, HF);
    k_out<<<NCRYS / 256, 256, 0, stream>>>(h2, out_W, out_b, out);

    (void)in_sizes; (void)n_in; (void)out_size; (void)ws_size;
}
